// HGTLink_73160472920276
// MI455X (gfx1250) — matrix operand order repaired and checked by reading, not compiled, not run on hardware
//
#include <hip/hip_runtime.h>
#include <hip/hip_bf16.h>

// ---------------------------------------------------------------------------
// MI455X (gfx1250) HGT link predictor.
// All dense GEMMs: v_wmma_f32_16x16x32_bf16 (fp32 accumulate).
// Weights are prepacked ONCE to bf16 in ISA B-fragment order; each GEMM block
// then DMAs the current col-tile into LDS with the Tensor Data Mover
// (tensor_load_to_lds + s_wait_tensorcnt), and lanes read their fragment as a
// single aligned 32B LDS load (2x ds_load_b128).
// Edge softmax/scatter use wave32 shuffles + global float atomics.
// ---------------------------------------------------------------------------

#ifndef USE_TDM
#define USE_TDM 1
#endif

typedef __attribute__((ext_vector_type(16))) __bf16    v16bf;
typedef __attribute__((ext_vector_type(8)))  float     v8f;
typedef __attribute__((ext_vector_type(4)))  unsigned  u32x4;
typedef __attribute__((ext_vector_type(8)))  unsigned  u32x8;

__device__ __forceinline__ float gelu_exact(float x) {
    return 0.5f * x * (1.0f + erff(x * 0.70710678118654752440f));
}

// Prepack W[K,M] (f32, row-major) -> bf16 in B-fragment order:
//   dst idx = ct*(K*16) + ks*512 + l*16 + j
//   with k = ks*32 + (j>>3)*16 + (l>>4)*8 + (j&7), col = ct*16 + (l&15).
// grid.y = batch (matrix stride K*M on both sides).
template<int K, int M>
__global__ __launch_bounds__(256)
void prepack_kernel(const float* __restrict__ W, __bf16* __restrict__ P)
{
    const int i = blockIdx.x * 256 + threadIdx.x;
    if (i >= K * M) return;
    const long base = (long)blockIdx.y * (K * M);
    const int ct = i / (K * 16);
    const int r  = i - ct * (K * 16);
    const int ks = r >> 9;
    const int r2 = r & 511;
    const int l  = r2 >> 4;
    const int j  = r2 & 15;
    const int k  = ks * 32 + ((j >> 3) << 4) + ((l >> 4) << 3) + (j & 7);
    const int n  = l & 15;
    P[base + i] = (__bf16)W[base + (size_t)k * M + ct * 16 + n];
}

// C[N,M] = EPI( PRE(A[N,K]) @ W[K,M] + bias )
// PRE: 0=none, 1=gelu      EPI: 0=none, 1=relu, 2=skip-mix (g*v + (1-g)*resid)
// 8 waves/block; wave owns a 16-row strip, iterates col tiles. Per col tile
// the packed weight chunk (K*32 bytes) is DMA'd into LDS by wave 0 via TDM.
template<int K, int M, int PRE, int EPI>
__global__ __launch_bounds__(256)
void wmma_gemm_kernel(const float* __restrict__ A, int lda, long bsA,
                      const __bf16* __restrict__ Wp, long bsW,
                      const float* __restrict__ bias,
                      float* __restrict__ C, int ldc, long bsC,
                      const float* __restrict__ resid,
                      const float* __restrict__ skipPtr,
                      int N)
{
    constexpr int KSTEPS = K / 32;
    constexpr int CTILES = M / 16;
    __shared__ alignas(32) __bf16 sWt[K * 16];   // one col tile, fragment order

    const int tid   = threadIdx.x;
    const int batch = blockIdx.y;
    A  += (long)batch * bsA;
    Wp += (long)batch * bsW;
    C  += (long)batch * bsC;

    const int lane = tid & 31;
    const int wave = tid >> 5;
    const int row0 = blockIdx.x * 128 + wave * 16;
    const bool active = row0 < N;     // wave-uniform: EXEC all-ones inside guards
    const int half = lane >> 4;
    const int mr   = lane & 15;

    // Load + convert A fragments for all K-steps (vectorized: 4x b128 per step).
    // ISA 7.12.2 16-bit A 16x32: element j -> k = (j>=8?16:0) + half*8 + (j&7).
    v16bf afr[KSTEPS];
    if (active) {
        const float* arow = A + (size_t)(row0 + mr) * lda;
#pragma unroll
        for (int ks = 0; ks < KSTEPS; ++ks) {
            const float4 a0 = *(const float4*)(arow + ks * 32 + half * 8);
            const float4 a1 = *(const float4*)(arow + ks * 32 + half * 8 + 4);
            const float4 a2 = *(const float4*)(arow + ks * 32 + 16 + half * 8);
            const float4 a3 = *(const float4*)(arow + ks * 32 + 16 + half * 8 + 4);
            float t[16] = {a0.x, a0.y, a0.z, a0.w, a1.x, a1.y, a1.z, a1.w,
                           a2.x, a2.y, a2.z, a2.w, a3.x, a3.y, a3.z, a3.w};
#pragma unroll
            for (int j = 0; j < 16; ++j) {
                float v = t[j];
                if constexpr (PRE == 1) v = gelu_exact(v);
                afr[ks][j] = (__bf16)v;
            }
        }
    }

    float g = 0.0f;
    if constexpr (EPI == 2) g = 1.0f / (1.0f + __expf(-skipPtr[0]));

    for (int ct = 0; ct < CTILES; ++ct) {
        const __bf16* src = Wp + (size_t)ct * (K * 16);
#if USE_TDM
        if (wave == 0) {
            // Tensor DMA descriptor (<=2D, groups 2/3 omitted):
            //   copy K*8 dwords (K*32 bytes) from `src` into LDS at sWt.
            const unsigned nw = (unsigned)(K * 8);        // 4-byte words
            const unsigned long long ga = (unsigned long long)(const void*)src;
            const unsigned ldsAddr = (unsigned)(unsigned long long)(uintptr_t)&sWt[0];
            u32x4 g0;
            g0[0] = 1u;                                   // count=1, user mode
            g0[1] = ldsAddr;                              // lds_addr [63:32]
            g0[2] = (unsigned)ga;                         // global_addr lo
            g0[3] = (unsigned)(ga >> 32) | (2u << 30);    // global_addr hi | type=2
            u32x8 g1;
            g1[0] = (2u << 16);                           // data_size=4B; wg_mask=0
            g1[1] = (nw & 0xFFFFu) << 16;                 // tensor_dim0 lo16
            g1[2] = (nw >> 16) | (1u << 16);              // tensor_dim0 hi | tensor_dim1=1
            g1[3] = (nw & 0xFFFFu) << 16;                 // tile_dim0 = nw
            g1[4] = 1u;                                   // tile_dim1=1, tile_dim2=0
            g1[5] = nw;                                   // tensor_dim0_stride lo32
            g1[6] = 0u;                                   // stride hi / dim1_stride lo
            g1[7] = 0u;
            asm volatile("tensor_load_to_lds %0, %1\n\t"
                         "s_wait_tensorcnt 0x0"
                         :: "s"(g0), "s"(g1) : "memory");
        }
#else
        // Fallback: contiguous vectorized copy (16B chunks).
        for (int s = tid; s < K * 2; s += 256)
            ((uint4*)sWt)[s] = ((const uint4*)src)[s];
#endif
        __syncthreads();

        if (active) {
            v8f acc = {0.f, 0.f, 0.f, 0.f, 0.f, 0.f, 0.f, 0.f};
#pragma unroll
            for (int ks = 0; ks < KSTEPS; ++ks) {
                const v16bf bfr = *(const v16bf*)(sWt + ks * 512 + lane * 16);
                acc = __builtin_amdgcn_wmma_f32_16x16x32_bf16(
                          false, afr[ks], false, bfr, (short)0, acc, false, false);
            }
            // Epilogue + store. C/D layout: row = row0 + half*8 + r, col = ct*16 + (lane&15).
            const int col = ct * 16 + (lane & 15);
            const float bv = bias ? bias[col] : 0.0f;    // hoisted: one load per tile
#pragma unroll
            for (int r = 0; r < 8; ++r) {
                const int row = row0 + half * 8 + r;
                float v = acc[r] + bv;
                if constexpr (EPI == 1) v = fmaxf(v, 0.0f);
                if constexpr (EPI == 2) v = g * v + (1.0f - g) * resid[(size_t)row * ldc + col];
                C[(size_t)row * ldc + col] = v;
            }
        }
        __syncthreads();   // WAR on sWt before next col tile
    }
}

// alpha_raw[e,h] = (q[dst[e],h,:] . kt[src[e],h,:]) * p_rel[h] / sqrt(64)
// plus segment-max into mEnc via monotonic uint encoding (init 0 == -inf).
__global__ __launch_bounds__(256)
void alpha_kernel(const int* __restrict__ edges, int E,
                  const float* __restrict__ q, const float* __restrict__ kt,
                  const float* __restrict__ p_rel,
                  float* __restrict__ alpha, unsigned* __restrict__ mEnc)
{
    const int wid  = (blockIdx.x * 256 + threadIdx.x) >> 5;
    const int lane = threadIdx.x & 31;
    if (wid >= E) return;
    const int src = edges[wid];
    const int dst = edges[E + wid];
    const float* qrow = q  + (size_t)dst * 192;
    const float* krow = kt + (size_t)src * 192;
    __builtin_prefetch(qrow, 0, 3);   // global_prefetch_b8
    __builtin_prefetch(krow, 0, 3);
    for (int h = 0; h < 3; ++h) {
        const float2 qv = *(const float2*)(qrow + h * 64 + lane * 2);
        const float2 kv = *(const float2*)(krow + h * 64 + lane * 2);
        float s = qv.x * kv.x + qv.y * kv.y;
        for (int off = 16; off > 0; off >>= 1) s += __shfl_xor(s, off, 32);
        if (lane == 0) {
            float a = s * p_rel[h] * 0.125f;   // 1/sqrt(64)
            alpha[(size_t)wid * 3 + h] = a;
            unsigned bits = __float_as_uint(a);
            unsigned enc  = (bits & 0x80000000u) ? ~bits : (bits | 0x80000000u);
            atomicMax(&mEnc[(size_t)dst * 3 + h], enc);
        }
    }
}

// alpha[e,h] = exp(alpha - m[dst,h]);  sbuf[dst,h] += alpha[e,h]
__global__ __launch_bounds__(256)
void exp_kernel(const int* __restrict__ edges, int E,
                float* __restrict__ alpha, const unsigned* __restrict__ mEnc,
                float* __restrict__ sbuf)
{
    const int idx = blockIdx.x * 256 + threadIdx.x;
    if (idx >= E * 3) return;
    const int e = idx / 3, h = idx - e * 3;
    const int dst = edges[E + e];
    const unsigned enc = mEnc[(size_t)dst * 3 + h];
    const float m = (enc & 0x80000000u) ? __uint_as_float(enc & 0x7fffffffu)
                                        : __uint_as_float(~enc);
    const float ex = __expf(alpha[idx] - m);
    alpha[idx] = ex;
    atomicAdd(&sbuf[(size_t)dst * 3 + h], ex);
}

// out[dst, c] += vt[src, c] * alpha[e,h] / (sbuf[dst,h] + 1e-16)
__global__ __launch_bounds__(256)
void scatter_kernel(const int* __restrict__ edges, int E,
                    const float* __restrict__ alpha, const float* __restrict__ sbuf,
                    const float* __restrict__ vt, float* __restrict__ out)
{
    const int idx = blockIdx.x * 256 + threadIdx.x;
    if (idx >= E * 192) return;
    const int e = idx / 192, c = idx - e * 192;
    const int h = c >> 6;
    const int src = edges[e];
    const int dst = edges[E + e];
    const float w = alpha[(size_t)e * 3 + h] / (sbuf[(size_t)dst * 3 + h] + 1e-16f);
    atomicAdd(&out[(size_t)dst * 192 + c], vt[(size_t)src * 192 + c] * w);
}

// out[el] = ec[i,:] . ek[j,:]   (64-dim dot, wave per label edge)
__global__ __launch_bounds__(256)
void eldot_kernel(const int* __restrict__ eli, int EL,
                  const float* __restrict__ ec, const float* __restrict__ ek,
                  float* __restrict__ out)
{
    const int wid  = (blockIdx.x * 256 + threadIdx.x) >> 5;
    const int lane = threadIdx.x & 31;
    if (wid >= EL) return;
    const int i = eli[wid];
    const int j = eli[EL + wid];
    const float2 a = *(const float2*)(ec + (size_t)i * 64 + lane * 2);
    const float2 b = *(const float2*)(ek + (size_t)j * 64 + lane * 2);
    float s = a.x * b.x + a.y * b.y;
    for (int off = 16; off > 0; off >>= 1) s += __shfl_xor(s, off, 32);
    if (lane == 0) out[wid] = s;
}

// ---------------------------------------------------------------------------

template<int K, int M, int PRE, int EPI>
static void launch_gemm(hipStream_t st, const float* A, int lda, long bsA,
                        const __bf16* Wp, long bsW, const float* bias,
                        float* C, int ldc, long bsC,
                        const float* resid, const float* skipPtr,
                        int N, int batch)
{
    dim3 grid((unsigned)((N + 127) / 128), (unsigned)batch, 1);
    wmma_gemm_kernel<K, M, PRE, EPI><<<grid, 256, 0, st>>>(
        A, lda, bsA, Wp, bsW, bias, C, ldc, bsC, resid, skipPtr, N);
}

extern "C" void kernel_launch(void* const* d_in, const int* in_sizes, int n_in,
                              void* d_out, int out_size, void* d_ws, size_t ws_size,
                              hipStream_t stream)
{
    (void)in_sizes; (void)n_in; (void)out_size; (void)ws_size;

    constexpr int N  = 50000;
    constexpr int E  = 250000;
    constexpr int EL = 200000;
    constexpr int HID = 192;

    const float* x_cpd = (const float*)d_in[0];
    const float* x_ko  = (const float*)d_in[1];
    const int*   e_c2k = (const int*)d_in[2];
    const int*   e_k2c = (const int*)d_in[3];
    const int*   eli   = (const int*)d_in[4];
    auto F = [&](int i) { return (const float*)d_in[i]; };
    // Param flattening (setup_inputs insertion order):
    //  5: in_cpd.w  6: in_cpd.b  7: in_ko.w  8: in_ko.b
    //  layer l base b = 9 + 24*l:
    //    cpd:  kw,kb,qw,qb,vw,vb,aw,ab,skip      -> b+0 .. b+8
    //    ko :  kw,kb,qw,qb,vw,vb,aw,ab,skip      -> b+9 .. b+17
    //    c2k:  a_rel,m_rel,p_rel                 -> b+18.. b+20
    //    k2c:  a_rel,m_rel,p_rel                 -> b+21.. b+23
    //  81: out.w  82: out.b

    const size_t NH = (size_t)N * HID;
    float* ws = (float*)d_ws;
    float* xc = ws;           float* xk = ws + NH;
    float* kc = ws + 2 * NH;  float* qc = ws + 3 * NH;  float* vc = ws + 4 * NH;
    float* kk = ws + 5 * NH;  float* qk = ws + 6 * NH;  float* vk = ws + 7 * NH;
    float* oc = ws + 8 * NH;  float* ok = ws + 9 * NH;
    float* ec = ws + 10 * NH;
    float* ek = ec + (size_t)N * 64;
    float* alpha = ek + (size_t)N * 64;
    float* sbuf  = alpha + (size_t)E * 3;
    unsigned* mEnc = (unsigned*)(sbuf + (size_t)N * 3);
    __bf16* pk = (__bf16*)(mEnc + (size_t)N * 3);   // prepacked weights (16B-aligned)
    auto nextPk = [&](size_t elems) { __bf16* p = pk; pk += elems; return p; };

    // ---- Prepack all weights to bf16 fragment order (runs every call; cheap) ----
    auto pp192x192 = [&](const float* W) {
        __bf16* p = nextPk(192 * 192);
        prepack_kernel<192, 192><<<dim3((192 * 192 + 255) / 256, 1), 256, 0, stream>>>(W, p);
        return p;
    };
    __bf16* p_in_c = nextPk(128 * 192);
    prepack_kernel<128, 192><<<dim3((128 * 192 + 255) / 256, 1), 256, 0, stream>>>(F(5), p_in_c);
    __bf16* p_in_k = nextPk(128 * 192);
    prepack_kernel<128, 192><<<dim3((128 * 192 + 255) / 256, 1), 256, 0, stream>>>(F(7), p_in_k);

    __bf16 *p_ck[3], *p_cq[3], *p_cv[3], *p_ca[3];
    __bf16 *p_kk[3], *p_kq[3], *p_kv[3], *p_ka[3];
    __bf16 *p_c2k_a[3], *p_c2k_m[3], *p_k2c_a[3], *p_k2c_m[3];
    for (int l = 0; l < 3; ++l) {
        const int b = 9 + l * 24;
        p_ck[l] = pp192x192(F(b + 0));
        p_cq[l] = pp192x192(F(b + 2));
        p_cv[l] = pp192x192(F(b + 4));
        p_ca[l] = pp192x192(F(b + 6));
        p_kk[l] = pp192x192(F(b + 9));
        p_kq[l] = pp192x192(F(b + 11));
        p_kv[l] = pp192x192(F(b + 13));
        p_ka[l] = pp192x192(F(b + 15));
        auto ppRel = [&](const float* W) {
            __bf16* p = nextPk(3 * 64 * 64);
            prepack_kernel<64, 64><<<dim3((64 * 64 + 255) / 256, 3), 256, 0, stream>>>(W, p);
            return p;
        };
        p_c2k_a[l] = ppRel(F(b + 18));
        p_c2k_m[l] = ppRel(F(b + 19));
        p_k2c_a[l] = ppRel(F(b + 21));
        p_k2c_m[l] = ppRel(F(b + 22));
    }
    __bf16* p_out = nextPk(192 * 64);
    prepack_kernel<192, 64><<<dim3((192 * 64 + 255) / 256, 1), 256, 0, stream>>>(F(81), p_out);

    const int edgeDotBlocks = (E * 32 + 255) / 256;
    const int expBlocks     = (E * 3 + 255) / 256;
    const int scatBlocks    = (E * 192 + 255) / 256;

    // Input projections: relu(x @ W + b)
    launch_gemm<128, 192, 0, 1>(stream, x_cpd, 128, 0, p_in_c, 0, F(6), xc, HID, 0, nullptr, nullptr, N, 1);
    launch_gemm<128, 192, 0, 1>(stream, x_ko,  128, 0, p_in_k, 0, F(8), xk, HID, 0, nullptr, nullptr, N, 1);

    for (int l = 0; l < 3; ++l) {
        const int b = 9 + l * 24;
        // K,Q,V projections
        launch_gemm<192, 192, 0, 0>(stream, xc, HID, 0, p_ck[l], 0, F(b+1),  kc, HID, 0, nullptr, nullptr, N, 1);
        launch_gemm<192, 192, 0, 0>(stream, xc, HID, 0, p_cq[l], 0, F(b+3),  qc, HID, 0, nullptr, nullptr, N, 1);
        launch_gemm<192, 192, 0, 0>(stream, xc, HID, 0, p_cv[l], 0, F(b+5),  vc, HID, 0, nullptr, nullptr, N, 1);
        launch_gemm<192, 192, 0, 0>(stream, xk, HID, 0, p_kk[l], 0, F(b+10), kk, HID, 0, nullptr, nullptr, N, 1);
        launch_gemm<192, 192, 0, 0>(stream, xk, HID, 0, p_kq[l], 0, F(b+12), qk, HID, 0, nullptr, nullptr, N, 1);
        launch_gemm<192, 192, 0, 0>(stream, xk, HID, 0, p_kv[l], 0, F(b+14), vk, HID, 0, nullptr, nullptr, N, 1);

        // --- relation c2k (src=cpd, dst=ko): per-head in-place transforms, then attention
        launch_gemm<64, 64, 0, 0>(stream, kc, HID, 64, p_c2k_a[l], 64*64, nullptr, kc, HID, 64, nullptr, nullptr, N, 3);
        launch_gemm<64, 64, 0, 0>(stream, vc, HID, 64, p_c2k_m[l], 64*64, nullptr, vc, HID, 64, nullptr, nullptr, N, 3);
        hipMemsetAsync(ok,   0, NH * sizeof(float), stream);
        hipMemsetAsync(sbuf, 0, (size_t)N * 3 * sizeof(float), stream);
        hipMemsetAsync(mEnc, 0, (size_t)N * 3 * sizeof(unsigned), stream);
        alpha_kernel<<<edgeDotBlocks, 256, 0, stream>>>(e_c2k, E, qk, kc, F(b+20), alpha, mEnc);
        exp_kernel  <<<expBlocks,     256, 0, stream>>>(e_c2k, E, alpha, mEnc, sbuf);
        scatter_kernel<<<scatBlocks,  256, 0, stream>>>(e_c2k, E, alpha, sbuf, vc, ok);

        // --- relation k2c (src=ko, dst=cpd)
        launch_gemm<64, 64, 0, 0>(stream, kk, HID, 64, p_k2c_a[l], 64*64, nullptr, kk, HID, 64, nullptr, nullptr, N, 3);
        launch_gemm<64, 64, 0, 0>(stream, vk, HID, 64, p_k2c_m[l], 64*64, nullptr, vk, HID, 64, nullptr, nullptr, N, 3);
        hipMemsetAsync(oc,   0, NH * sizeof(float), stream);
        hipMemsetAsync(sbuf, 0, (size_t)N * 3 * sizeof(float), stream);
        hipMemsetAsync(mEnc, 0, (size_t)N * 3 * sizeof(unsigned), stream);
        alpha_kernel<<<edgeDotBlocks, 256, 0, stream>>>(e_k2c, E, qc, kk, F(b+23), alpha, mEnc);
        exp_kernel  <<<expBlocks,     256, 0, stream>>>(e_k2c, E, alpha, mEnc, sbuf);
        scatter_kernel<<<scatBlocks,  256, 0, stream>>>(e_k2c, E, alpha, sbuf, vk, oc);

        // Updates: x = g * (gelu(out) @ aw + ab) + (1-g) * x
        launch_gemm<192, 192, 1, 2>(stream, oc, HID, 0, p_ca[l], 0, F(b+7),  xc, HID, 0, xc, F(b+8),  N, 1);
        launch_gemm<192, 192, 1, 2>(stream, ok, HID, 0, p_ka[l], 0, F(b+16), xk, HID, 0, xk, F(b+17), N, 1);
    }

    // Output projection + edge-label dot
    launch_gemm<192, 64, 0, 0>(stream, xc, HID, 0, p_out, 0, F(82), ec, 64, 0, nullptr, nullptr, N, 1);
    launch_gemm<192, 64, 0, 0>(stream, xk, HID, 0, p_out, 0, F(82), ek, 64, 0, nullptr, nullptr, N, 1);
    eldot_kernel<<<(EL * 32 + 255) / 256, 256, 0, stream>>>(eli, EL, ec, ek, (float*)d_out);
}
